// TemporalContextualGating_39848706573762
// MI455X (gfx1250) — matrix operand order repaired and checked by reading, not compiled, not run on hardware
//
#include <hip/hip_runtime.h>
#include <hip/hip_bf16.h>

typedef __attribute__((ext_vector_type(16))) __bf16 v16bf;
typedef __attribute__((ext_vector_type(8)))  float  v8f;
typedef __attribute__((ext_vector_type(4)))  unsigned int v4u;

#define D_MODEL 1024
#define SEQ_L   4096
#define DC_     256
#define KPAT    8
#define KTOT    7168   // 7 * 1024 folded-conv reduction dim
#define NPERIOD 112    // 2 chunks (32K-slice each) per LDS stage period

// ---- f32 -> bf16 round-to-nearest-even (prep kernel only) ----
__device__ __forceinline__ unsigned short f2bf(float f) {
  union { float f; unsigned u; } v; v.f = f;
  unsigned r = v.u + 0x7FFFu + ((v.u >> 16) & 1u);
  return (unsigned short)(r >> 16);
}

// generic pointer -> raw LDS byte offset (for DS inline asm operands)
__device__ __forceinline__ unsigned lds_off(const void* p) {
  return (unsigned)(unsigned long long)(__attribute__((address_space(3))) const void*)p;
}

// =====================================================================
// Prep: fold depthwise conv weights into the context GEMM weight,
// transposed [DC][7*D] in bf16 (L2-resident, 3.67 MB); also produce
// L2-normalized centroids and the effective context bias.
// =====================================================================
__global__ void tcg_prep(const float* __restrict__ w3, const float* __restrict__ b3,
                         const float* __restrict__ w7, const float* __restrict__ b7,
                         const float* __restrict__ W_ctx, const float* __restrict__ b_ctx,
                         const float* __restrict__ centroids,
                         unsigned short* __restrict__ WT,
                         float* __restrict__ centn, float* __restrict__ bce) {
  __shared__ float wsum[8];
  const int blk = blockIdx.x;
  const int t   = threadIdx.x;   // 0..255
  if (blk < KTOT) {
    const int s = blk >> 10;     // shift index 0..6  (shift = s-3)
    const int d = blk & 1023;
    const float w7v = w7[d * 7 + s];
    float w3v = 0.0f;
    if (s >= 2 && s <= 4) w3v = w3[d * 3 + (s - 2)];
    const float v = w3v * W_ctx[t * 2048 + d] + w7v * W_ctx[t * 2048 + 1024 + d];
    WT[(size_t)t * KTOT + blk] = f2bf(v);
  } else if (blk < KTOT + KPAT) {
    const int k = blk - KTOT;
    const float v = centroids[k * DC_ + t];
    float ss = v * v;
    #pragma unroll
    for (int m = 16; m >= 1; m >>= 1) ss += __shfl_xor(ss, m, 32);
    const int lane = t & 31, wv = t >> 5;
    if (lane == 0) wsum[wv] = ss;
    __syncthreads();
    float tot = 0.0f;
    #pragma unroll
    for (int i = 0; i < 8; ++i) tot += wsum[i];
    centn[k * DC_ + t] = v / fmaxf(sqrtf(tot), 1e-12f);
  } else {
    float acc = b_ctx[t];
    for (int d = 0; d < 1024; ++d)
      acc += b3[d] * W_ctx[t * 2048 + d] + b7[d] * W_ctx[t * 2048 + 1024 + d];
    bce[t] = acc;
  }
}

// =====================================================================
// Main fused kernel. One workgroup = 64 (b,l)-rows x DC=256 cols.
// 8 waves laid out 4(M) x 2(N); each wave: 16 rows x 128 cols.
// B is double-buffered through LDS (2 chunks = 32KB per period).
// Fragment DS loads + dscnt waits are hand-scheduled via inline asm so
// 2-3 fragments stay in flight ahead of the WMMA chain.
// =====================================================================
__global__ void __launch_bounds__(256, 1)
tcg_main(const float* __restrict__ x,
         const unsigned short* __restrict__ WT,
         const float* __restrict__ bce,
         const float* __restrict__ centn,
         const float* __restrict__ rscale,
         const float* __restrict__ mode_table,
         const float* __restrict__ gammap,
         float* __restrict__ out) {
  __shared__ __align__(16) unsigned char smem[65536];
  unsigned short* Bst = (unsigned short*)smem;  // [2 buf][2 chunk][256 n][32 k]
  float* cbuf = (float*)smem;                   // [64][256] epilogue alias

  const int t    = threadIdx.x;
  const int lane = t & 31;
  const int wv   = t >> 5;
  const int mw   = wv & 3;           // 0..3 : M tile
  const int nw   = wv >> 2;          // 0..1 : N tile
  const int R0   = blockIdx.x * 64;  // global row base over B*L
  const int b    = R0 >> 12;
  const int l0   = R0 & 4095;
  const bool edge = (l0 == 0) || (l0 == SEQ_L - 64);

  v8f acc[8];
  #pragma unroll
  for (int i = 0; i < 8; ++i)
    #pragma unroll
    for (int j = 0; j < 8; ++j) acc[i][j] = 0.0f;

  // A 16x32 bf16 lane layout: lanes 0-15 -> M=lane, K 0-7 & 16-23;
  //                           lanes 16-31 -> M=lane-16, K 8-15 & 24-31.
  const int arow  = mw * 16 + (lane & 15);
  const int koffA = (lane < 16) ? 0 : 8;
  // B 32x16 lane layout: lanes 0-15 -> K 0-7 & 16-23; lanes 16-31 -> K 8-15 & 24-31.
  const int koffB = (lane < 16) ? 0 : 8;
  const int nrow  = nw * 128 + (lane & 15);

  // Cooperative B staging map: unit u in [0,2048): 16B per unit.
  #pragma unroll
  for (int i = 0; i < 8; ++i) {
    const int u = t + 256 * i;
    const int cpart = u >> 10, n = (u >> 2) & 255, c16 = u & 3;
    const uint4 d = *(const uint4*)(WT + (size_t)n * KTOT + cpart * 32 + c16 * 8);
    *(uint4*)(Bst + cpart * 8192 + n * 32 + c16 * 8) = d;
  }
  __syncthreads();

  // Per-wave LDS byte base for fragment reads (buffer select added per period)
  const unsigned ldsbase = lds_off(Bst) + (unsigned)(nrow * 64 + koffB * 2);

  union Au { v16bf v; unsigned int u[8]; };
  union Bu { v16bf v; v4u q[2]; };

#define LOADB_ASM(d0, d1, offc)                                        \
  asm volatile("ds_load_b128 %0, %2 offset:%c3\n\t"                    \
               "ds_load_b128 %1, %2 offset:%c4"                        \
               : "=v"(d0), "=v"(d1)                                    \
               : "v"(ba), "i"(offc), "i"((offc) + 32))
#define WAITB(d0, d1, cnt)                                             \
  asm volatile("s_wait_dscnt %c2" : "+v"(d0), "+v"(d1) : "i"(cnt))
#define WMMAC(nt, qa, qb, AV)                                          \
  { Bu Bt; Bt.q[0] = (qa); Bt.q[1] = (qb);                             \
    acc[nt] = __builtin_amdgcn_wmma_f32_16x16x32_bf16(                 \
        false, (AV), false, Bt.v, (short)0, acc[nt], false, false); }
// Depth-2 fragment pipeline with manual dscnt waits (2 loads per frag).
#define GEMM_HALF(HOFF, AV)                                            \
  { v4u b0a, b0b, b1a, b1b, b2a, b2b;                                  \
    LOADB_ASM(b0a, b0b, (HOFF) + 0 * 1024);                            \
    LOADB_ASM(b1a, b1b, (HOFF) + 1 * 1024);                            \
    LOADB_ASM(b2a, b2b, (HOFF) + 2 * 1024);                            \
    WAITB(b0a, b0b, 4); WMMAC(0, b0a, b0b, AV);                        \
    LOADB_ASM(b0a, b0b, (HOFF) + 3 * 1024);                            \
    WAITB(b1a, b1b, 4); WMMAC(1, b1a, b1b, AV);                        \
    LOADB_ASM(b1a, b1b, (HOFF) + 4 * 1024);                            \
    WAITB(b2a, b2b, 4); WMMAC(2, b2a, b2b, AV);                        \
    LOADB_ASM(b2a, b2b, (HOFF) + 5 * 1024);                            \
    WAITB(b0a, b0b, 4); WMMAC(3, b0a, b0b, AV);                        \
    LOADB_ASM(b0a, b0b, (HOFF) + 6 * 1024);                            \
    WAITB(b1a, b1b, 4); WMMAC(4, b1a, b1b, AV);                        \
    LOADB_ASM(b1a, b1b, (HOFF) + 7 * 1024);                            \
    WAITB(b2a, b2b, 4); WMMAC(5, b2a, b2b, AV);                        \
    WAITB(b0a, b0b, 2); WMMAC(6, b0a, b0b, AV);                        \
    WAITB(b1a, b1b, 0); WMMAC(7, b1a, b1b, AV); }

#define CVT_A(Adst, pA)                                                \
  { const float4 f0 = *(const float4*)(pA);                            \
    const float4 f1 = *(const float4*)((pA) + 4);                      \
    const float4 f2 = *(const float4*)((pA) + 16);                     \
    const float4 f3 = *(const float4*)((pA) + 20);                     \
    Adst.v[0]=(__bf16)f0.x; Adst.v[1]=(__bf16)f0.y;                    \
    Adst.v[2]=(__bf16)f0.z; Adst.v[3]=(__bf16)f0.w;                    \
    Adst.v[4]=(__bf16)f1.x; Adst.v[5]=(__bf16)f1.y;                    \
    Adst.v[6]=(__bf16)f1.z; Adst.v[7]=(__bf16)f1.w;                    \
    Adst.v[8]=(__bf16)f2.x; Adst.v[9]=(__bf16)f2.y;                    \
    Adst.v[10]=(__bf16)f2.z; Adst.v[11]=(__bf16)f2.w;                  \
    Adst.v[12]=(__bf16)f3.x; Adst.v[13]=(__bf16)f3.y;                  \
    Adst.v[14]=(__bf16)f3.z; Adst.v[15]=(__bf16)f3.w;                  \
    _Pragma("unroll")                                                  \
    for (int i_ = 0; i_ < 8; ++i_) Adst.u[i_] = valid ? Adst.u[i_] : 0u; }

  for (int p = 0; p < NPERIOD; ++p) {
    const int cur = p & 1, nxt = cur ^ 1;

    // ---- issue next period's global loads (latency hidden by WMMAs) ----
    uint4 g[8];
    if (p < NPERIOD - 1) {
      const int cbase = (p + 1) * 2;
      #pragma unroll
      for (int i = 0; i < 8; ++i) {
        const int u = t + 256 * i;
        const int cpart = u >> 10, n = (u >> 2) & 255, c16 = u & 3;
        g[i] = *(const uint4*)(WT + (size_t)n * KTOT + (cbase + cpart) * 32 + c16 * 8);
      }
    }

    // ---- A row for this period (both chunks share the same shift s) ----
    const int s   = (2 * p) >> 5;
    const int kc0 = (2 * p) & 31;
    const int lb  = l0 + arow + s - 3;                     // SAME padding
    const bool valid = !edge || ((lb >= 0) && (lb < SEQ_L));
    const int lbc = valid ? lb : 0;                        // clamped, safe addr
    const float* xr = x + ((size_t)b * SEQ_L + lbc) * D_MODEL + kc0 * 32 + koffA;

    Au A0, A1;
    CVT_A(A0, xr);
    CVT_A(A1, xr + 32);

    const unsigned ba = ldsbase + (unsigned)cur * 32768u;
    GEMM_HALF(0,     A0.v);
    GEMM_HALF(16384, A1.v);

    // ---- commit prefetched data into the other buffer ----
    if (p < NPERIOD - 1) {
      unsigned short* db = Bst + nxt * 16384;
      #pragma unroll
      for (int i = 0; i < 8; ++i) {
        const int u = t + 256 * i;
        const int cpart = u >> 10, n = (u >> 2) & 255, c16 = u & 3;
        *(uint4*)(db + cpart * 8192 + n * 32 + c16 * 8) = g[i];
      }
    }
    __syncthreads();
  }
#undef LOADB_ASM
#undef WAITB
#undef WMMAC
#undef GEMM_HALF
#undef CVT_A

  // ---- epilogue phase 1: bias + exact GELU -> LDS (cbuf aliases Bst) ----
  // C/D layout: VGPR j -> M = j + (lane>=16 ? 8 : 0), N = lane&15.
  const int rbase = mw * 16 + ((lane >= 16) ? 8 : 0);
  #pragma unroll
  for (int nt = 0; nt < 8; ++nt) {
    const int col = nw * 128 + nt * 16 + (lane & 15);
    const float bc = bce[col];
    #pragma unroll
    for (int j = 0; j < 8; ++j) {
      const float v = acc[nt][j] + bc;
      const float gl = 0.5f * v * (1.0f + erff(v * 0.70710678118654752f));
      cbuf[(rbase + j) * 256 + col] = gl;
    }
  }
  __syncthreads();

  // ---- epilogue phase 2: per row (4 lanes/row): l2norm, cosine logits,
  //      softmax(K=8), gated output ----
  const int r = t >> 2;          // 0..63
  const int q = t & 3;           // 64-col slice
  const float* crow = &cbuf[r * 256 + q * 64];
  float ss = 0.0f;
  float lg[8];
  #pragma unroll
  for (int k = 0; k < 8; ++k) lg[k] = 0.0f;
  for (int c = 0; c < 64; ++c) {
    const float cv = crow[c];
    ss += cv * cv;
    const int gc = q * 64 + c;
    #pragma unroll
    for (int k = 0; k < 8; ++k) lg[k] += cv * centn[k * 256 + gc];
  }
  ss += __shfl_xor(ss, 1, 32);
  ss += __shfl_xor(ss, 2, 32);
  const float inv = 1.0f / fmaxf(sqrtf(ss), 1e-12f);
  const float rs  = rscale[0] * inv;
  float mx = -1e30f;
  #pragma unroll
  for (int k = 0; k < 8; ++k) {
    lg[k] += __shfl_xor(lg[k], 1, 32);
    lg[k] += __shfl_xor(lg[k], 2, 32);
    lg[k] *= rs;
    mx = fmaxf(mx, lg[k]);
  }
  float se = 0.0f, pk[8];
  #pragma unroll
  for (int k = 0; k < 8; ++k) { pk[k] = __expf(lg[k] - mx); se += pk[k]; }
  const float scale = gammap[0] / se;   // fold gamma into probabilities
  #pragma unroll
  for (int k = 0; k < 8; ++k) pk[k] *= scale;

  const size_t grow = (size_t)(R0 + r) * D_MODEL;
  const float* xr2  = x + grow;
  float* outr       = out + grow;
  for (int d0 = q * 256; d0 < q * 256 + 256; d0 += 4) {
    const float4 xv = *(const float4*)(xr2 + d0);
    float m0 = 0.f, m1 = 0.f, m2 = 0.f, m3 = 0.f;
    #pragma unroll
    for (int k = 0; k < 8; ++k) {
      const float4 mv = *(const float4*)(mode_table + k * 1024 + d0);
      m0 += pk[k] * mv.x; m1 += pk[k] * mv.y; m2 += pk[k] * mv.z; m3 += pk[k] * mv.w;
    }
    float4 ov;
    ov.x = xv.x * (1.0f + m0);
    ov.y = xv.y * (1.0f + m1);
    ov.z = xv.z * (1.0f + m2);
    ov.w = xv.w * (1.0f + m3);
    *(float4*)(outr + d0) = ov;
  }
}

extern "C" void kernel_launch(void* const* d_in, const int* in_sizes, int n_in,
                              void* d_out, int out_size, void* d_ws, size_t ws_size,
                              hipStream_t stream) {
  const float* x         = (const float*)d_in[0];
  const float* w3        = (const float*)d_in[1];
  const float* b3        = (const float*)d_in[2];
  const float* w7        = (const float*)d_in[3];
  const float* b7        = (const float*)d_in[4];
  const float* W_ctx     = (const float*)d_in[5];
  const float* b_ctx     = (const float*)d_in[6];
  const float* centroids = (const float*)d_in[7];
  const float* rscale    = (const float*)d_in[8];
  const float* mode_tab  = (const float*)d_in[9];
  const float* gamma     = (const float*)d_in[10];
  float* out = (float*)d_out;

  // ws layout: WT bf16 [256][7168] | centn f32 [8][256] | bce f32 [256]
  unsigned short* WT = (unsigned short*)d_ws;
  const size_t WT_BYTES = (size_t)DC_ * KTOT * sizeof(unsigned short); // 3,670,016
  float* centn = (float*)((char*)d_ws + WT_BYTES);
  float* bce   = (float*)((char*)d_ws + WT_BYTES + KPAT * DC_ * sizeof(float));

  tcg_prep<<<KTOT + KPAT + 1, 256, 0, stream>>>(w3, b3, w7, b7, W_ctx, b_ctx,
                                                centroids, WT, centn, bce);
  tcg_main<<<512, 256, 0, stream>>>(x, WT, bce, centn, rscale, mode_tab, gamma, out);
}
